// CausalSelfAttention_18691697672467
// MI455X (gfx1250) — matrix operand order repaired and checked by reading, not compiled, not run on hardware
//
#include <hip/hip_runtime.h>
#include <hip/hip_bf16.h>

// Problem constants (match reference)
#define NEMB  1024
#define NH    16
#define TSEQ  2048
#define BATCH 4
#define HDIM  64
#define MTOT  (BATCH * TSEQ)   // 8192

typedef __attribute__((ext_vector_type(16))) _Float16 v16h;
typedef __attribute__((ext_vector_type(8)))  float    v8f;

union AF { v16h v; float4 f4[2]; _Float16 h[16]; };
union H8 { float4 f4; _Float16 h[8]; };
union H4 { float2 f2; _Float16 h[4]; };

static __device__ __forceinline__ v8f vzero8() {
  v8f z;
  #pragma unroll
  for (int i = 0; i < 8; ++i) z[i] = 0.0f;
  return z;
}

static __device__ __forceinline__ v8f wmma16(const AF& a, const AF& b, v8f c) {
  // D = A(16x32 f16) * B(32x16 f16) + C(16x16 f32)
  return __builtin_amdgcn_wmma_f32_16x16x32_f16(false, a.v, false, b.v,
                                                (short)0, c, false, false);
}

// ---------------------------------------------------------------------------
// fp32 -> f16 conversion, 4 elements per thread
// ---------------------------------------------------------------------------
__global__ void cvt_f32_f16_k(const float* __restrict__ in,
                              _Float16* __restrict__ out, int n4) {
  int i = blockIdx.x * blockDim.x + threadIdx.x;
  if (i < n4) {
    float4 v = ((const float4*)in)[i];
    H4 u;
    u.h[0] = (_Float16)v.x; u.h[1] = (_Float16)v.y;
    u.h[2] = (_Float16)v.z; u.h[3] = (_Float16)v.w;
    ((float2*)out)[i] = u.f2;
  }
}

// ---------------------------------------------------------------------------
// GEMM: D[m,n] = sum_c X[m,c] * W[n,c]   (out = x @ W^T)
// One wave computes a 32(M) x 64(N) tile: 8 accumulators, B reused across two
// A tiles -> 8 WMMAs per 12 b128 loads per K-step of 32.
// mode 0: f16 out, Q/K layout [B,H,T,64]
// mode 1: f16 out, V transposed  [B,H,64,T]
// mode 2: f32 out, plain row-major [M, N]
// ---------------------------------------------------------------------------
__global__ __launch_bounds__(32)
void gemm_xw_k(const _Float16* __restrict__ X, const _Float16* __restrict__ W,
               void* __restrict__ outp, int mode) {
  const int lane   = threadIdx.x;
  const int lane16 = lane & 15;
  const int hf     = lane >> 4;           // half-wave select
  const int mbase  = blockIdx.x * 32;
  const int nbase  = blockIdx.y * 64;

  const _Float16* arow0 = X + (size_t)(mbase + lane16) * NEMB;
  const _Float16* arow1 = X + (size_t)(mbase + 16 + lane16) * NEMB;

  v8f acc[2][4];
  #pragma unroll
  for (int i = 0; i < 2; ++i)
    #pragma unroll
    for (int j = 0; j < 4; ++j) acc[i][j] = vzero8();

  for (int k = 0; k < NEMB; k += 32) {
    // A 16x32 layout: lane holds row m=lane16, K = {hf*8..+7, 16+hf*8..+7}
    AF a0, a1;
    a0.f4[0] = *(const float4*)(arow0 + k + hf * 8);
    a0.f4[1] = *(const float4*)(arow0 + k + 16 + hf * 8);
    a1.f4[0] = *(const float4*)(arow1 + k + hf * 8);
    a1.f4[1] = *(const float4*)(arow1 + k + 16 + hf * 8);
    #pragma unroll
    for (int j = 0; j < 4; ++j) {
      // B 32x16 layout: lane holds col n=lane16, K = {hf*8..+7, 16+hf*8..+7}
      const _Float16* brow =
          W + (size_t)(nbase + j * 16 + lane16) * NEMB + k + hf * 8;
      AF b;
      b.f4[0] = *(const float4*)(brow);
      b.f4[1] = *(const float4*)(brow + 16);
      acc[0][j] = wmma16(a0, b, acc[0][j]);
      acc[1][j] = wmma16(a1, b, acc[1][j]);
    }
  }

  // C/D layout: VGPR r <-> row M = r + 8*hf ; col N = lane16
  #pragma unroll
  for (int i = 0; i < 2; ++i) {
    const int mb   = mbase + i * 16;
    const int bidx = mb / TSEQ;       // 32-row tile never crosses a batch row
    const int tloc = mb % TSEQ;

    if (mode == 2) {
      float* out = (float*)outp;
      #pragma unroll
      for (int j = 0; j < 4; ++j) {
        const int n = nbase + j * 16 + lane16;
        #pragma unroll
        for (int r = 0; r < 8; ++r)
          out[(size_t)(mb + r + 8 * hf) * NEMB + n] = acc[i][j][r];
      }
    } else if (mode == 0) {
      _Float16* out = (_Float16*)outp;
      #pragma unroll
      for (int j = 0; j < 4; ++j) {
        const int n  = nbase + j * 16 + lane16;
        const int hh = n >> 6, hd = n & 63;
        const size_t base = ((size_t)(bidx * NH + hh) * TSEQ) * HDIM + hd;
        #pragma unroll
        for (int r = 0; r < 8; ++r)
          out[base + (size_t)(tloc + r + 8 * hf) * HDIM] = (_Float16)acc[i][j][r];
      }
    } else { // mode 1: V transposed [B,H,64,T]; lane's 8 rows = 8 consecutive t
      _Float16* out = (_Float16*)outp;
      #pragma unroll
      for (int j = 0; j < 4; ++j) {
        const int n  = nbase + j * 16 + lane16;
        const int hh = n >> 6, hd = n & 63;
        H8 pk;
        #pragma unroll
        for (int r = 0; r < 8; ++r) pk.h[r] = (_Float16)acc[i][j][r];
        const size_t off =
            ((size_t)(bidx * NH + hh) * HDIM + hd) * TSEQ + tloc + 8 * hf;
        *(float4*)(out + off) = pk.f4;
      }
    }
  }
}

// ---------------------------------------------------------------------------
// Flash attention: one wave per (b, h, 16-row q tile). K loop step 32.
// Q,K: [B,H,T,64] f16 ; Vt: [B,H,64,T] f16 ; O: [B,T,1024] f16
// ---------------------------------------------------------------------------
__global__ __launch_bounds__(32)
void attn_k(const _Float16* __restrict__ Q, const _Float16* __restrict__ K,
            const _Float16* __restrict__ Vt, _Float16* __restrict__ O) {
  __shared__ __align__(16) _Float16 Plds[16 * 32];

  const int lane   = threadIdx.x;
  const int lane16 = lane & 15;
  const int hf     = lane >> 4;
  const int qbase  = blockIdx.x * 16;
  const int h      = blockIdx.y;
  const int b      = blockIdx.z;
  const size_t bh  = (size_t)(b * NH + h);

  const _Float16* Qb = Q  + bh * TSEQ * HDIM;
  const _Float16* Kb = K  + bh * TSEQ * HDIM;
  const _Float16* Vb = Vt + bh * HDIM * TSEQ;

  // Load Q tile once: two A matrices (d = 0..31 and d = 32..63)
  AF qa[2];
  {
    const _Float16* qrow = Qb + (size_t)(qbase + lane16) * HDIM;
    qa[0].f4[0] = *(const float4*)(qrow + hf * 8);
    qa[0].f4[1] = *(const float4*)(qrow + 16 + hf * 8);
    qa[1].f4[0] = *(const float4*)(qrow + 32 + hf * 8);
    qa[1].f4[1] = *(const float4*)(qrow + 48 + hf * 8);
  }

  float mrow[8], lrow[8];
  v8f oacc[4];
  #pragma unroll
  for (int r = 0; r < 8; ++r) { mrow[r] = -3.0e30f; lrow[r] = 0.0f; }
  #pragma unroll
  for (int d = 0; d < 4; ++d) oacc[d] = vzero8();

  for (int kt = 0; kt <= qbase + 15; kt += 32) {
    // ---- scores S (16q x 32k), f32 accum, 4 WMMAs ----
    v8f s0 = vzero8(), s1 = vzero8();
    #pragma unroll
    for (int ds = 0; ds < 2; ++ds) {
      const _Float16* k0 = Kb + (size_t)(kt + lane16) * HDIM + ds * 32 + hf * 8;
      const _Float16* k1 = Kb + (size_t)(kt + 16 + lane16) * HDIM + ds * 32 + hf * 8;
      AF b0, b1;
      b0.f4[0] = *(const float4*)(k0); b0.f4[1] = *(const float4*)(k0 + 16);
      b1.f4[0] = *(const float4*)(k1); b1.f4[1] = *(const float4*)(k1 + 16);
      s0 = wmma16(qa[ds], b0, s0);
      s1 = wmma16(qa[ds], b1, s1);
    }

    // ---- scale, causal mask, online softmax (row = r + 8*hf) ----
    #pragma unroll
    for (int r = 0; r < 8; ++r) {
      const int qrow = qbase + r + 8 * hf;
      float v0 = s0[r] * 0.125f;                 // 1/sqrt(64)
      float v1 = s1[r] * 0.125f;
      if (kt + lane16 > qrow)      v0 = -3.0e30f;
      if (kt + 16 + lane16 > qrow) v1 = -3.0e30f;
      float mx = fmaxf(v0, v1);
      #pragma unroll
      for (int off = 1; off < 16; off <<= 1)
        mx = fmaxf(mx, __shfl_xor(mx, off, 16));
      const float mnew = fmaxf(mrow[r], mx);
      const float sc   = __expf(mrow[r] - mnew);
      const float p0   = __expf(v0 - mnew);
      const float p1   = __expf(v1 - mnew);
      float rs = p0 + p1;
      #pragma unroll
      for (int off = 1; off < 16; off <<= 1)
        rs += __shfl_xor(rs, off, 16);
      lrow[r] = lrow[r] * sc + rs;
      mrow[r] = mnew;
      #pragma unroll
      for (int d = 0; d < 4; ++d) oacc[d][r] *= sc;
      // stage P (C/D layout -> row-major 16x32 f16 in LDS)
      const int prow = r + 8 * hf;
      Plds[prow * 32 + lane16]      = (_Float16)p0;
      Plds[prow * 32 + 16 + lane16] = (_Float16)p1;
    }

    // wave-internal LDS visibility: wait for ds stores, then reload as A-layout
    asm volatile("s_wait_dscnt 0" ::: "memory");
    AF pa;
    pa.f4[0] = *(const float4*)(&Plds[lane16 * 32 + hf * 8]);
    pa.f4[1] = *(const float4*)(&Plds[lane16 * 32 + 16 + hf * 8]);
    asm volatile("" ::: "memory");

    // ---- O += P(16x32) x V(32x16) per 16-wide d tile, 4 WMMAs ----
    #pragma unroll
    for (int d = 0; d < 4; ++d) {
      const _Float16* vp = Vb + (size_t)(d * 16 + lane16) * TSEQ + kt + hf * 8;
      AF bv;
      bv.f4[0] = *(const float4*)(vp);
      bv.f4[1] = *(const float4*)(vp + 16);
      oacc[d] = wmma16(pa, bv, oacc[d]);
    }
  }

  // ---- normalize and store O as [B,T,1024] f16 ----
  #pragma unroll
  for (int r = 0; r < 8; ++r) {
    const float inv = 1.0f / lrow[r];
    const int t = qbase + r + 8 * hf;
    _Float16* orow = O + ((size_t)b * TSEQ + t) * NEMB + h * HDIM;
    #pragma unroll
    for (int d = 0; d < 4; ++d)
      orow[d * 16 + lane16] = (_Float16)(oacc[d][r] * inv);
  }
}

// ---------------------------------------------------------------------------
extern "C" void kernel_launch(void* const* d_in, const int* in_sizes, int n_in,
                              void* d_out, int out_size, void* d_ws, size_t ws_size,
                              hipStream_t stream) {
  const float* x  = (const float*)d_in[0];
  const float* wq = (const float*)d_in[1];
  const float* wk = (const float*)d_in[2];
  const float* wv = (const float*)d_in[3];
  const float* wo = (const float*)d_in[4];

  char* ws = (char*)d_ws;
  size_t off = 0;
  auto alloc = [&](size_t bytes) -> void* {
    void* p = ws + off;
    off += (bytes + 255) & ~(size_t)255;
    return p;
  };
  const size_t xhB = (size_t)MTOT * NEMB * 2;
  const size_t wB  = (size_t)NEMB * NEMB * 2;
  const size_t qB  = (size_t)BATCH * NH * TSEQ * HDIM * 2;

  _Float16* xh  = (_Float16*)alloc(xhB);
  _Float16* wqh = (_Float16*)alloc(wB);
  _Float16* wkh = (_Float16*)alloc(wB);
  _Float16* wvh = (_Float16*)alloc(wB);
  _Float16* woh = (_Float16*)alloc(wB);
  _Float16* Qh  = (_Float16*)alloc(qB);
  _Float16* Kh  = (_Float16*)alloc(qB);
  _Float16* Vth = (_Float16*)alloc(qB);
  _Float16* Oh  = (_Float16*)alloc(xhB);

  // 1) fp32 -> f16 conversions
  {
    const int nx4 = (MTOT * NEMB) / 4;      // 2,097,152
    const int nw4 = (NEMB * NEMB) / 4;      // 262,144
    cvt_f32_f16_k<<<(nx4 + 255) / 256, 256, 0, stream>>>(x,  xh,  nx4);
    cvt_f32_f16_k<<<(nw4 + 255) / 256, 256, 0, stream>>>(wq, wqh, nw4);
    cvt_f32_f16_k<<<(nw4 + 255) / 256, 256, 0, stream>>>(wk, wkh, nw4);
    cvt_f32_f16_k<<<(nw4 + 255) / 256, 256, 0, stream>>>(wv, wvh, nw4);
    cvt_f32_f16_k<<<(nw4 + 255) / 256, 256, 0, stream>>>(wo, woh, nw4);
  }

  // 2) Q/K/V projections (WMMA GEMMs), 32x64 tile per wave
  dim3 gw(MTOT / 32, NEMB / 64, 1);        // (256, 16)
  gemm_xw_k<<<gw, 32, 0, stream>>>(xh, wqh, Qh,  0);
  gemm_xw_k<<<gw, 32, 0, stream>>>(xh, wkh, Kh,  0);
  gemm_xw_k<<<gw, 32, 0, stream>>>(xh, wvh, Vth, 1);

  // 3) flash attention
  attn_k<<<dim3(TSEQ / 16, NH, BATCH), 32, 0, stream>>>(Qh, Kh, Vth, Oh);

  // 4) output projection -> fp32 d_out
  gemm_xw_k<<<gw, 32, 0, stream>>>(Oh, woh, d_out, 2);
}
